// SSMnet_36979668418742
// MI455X (gfx1250) — matrix operand order repaired and checked by reading, not compiled, not run on hardware
//
#include <hip/hip_runtime.h>
#include <hip/hip_bf16.h>
#include <math.h>

typedef __attribute__((ext_vector_type(16))) _Float16 v16h;
typedef __attribute__((ext_vector_type(8)))  float    v8f;
typedef __attribute__((ext_vector_type(4)))  int      v4i;

#define AS1 __attribute__((address_space(1)))
#define AS3 __attribute__((address_space(3)))

#if defined(__has_builtin)
#if __has_builtin(__builtin_amdgcn_global_load_async_to_lds_b128)
#define HAVE_ASYNC_LDS 1
#endif
#endif
#ifndef HAVE_ASYNC_LDS
#define HAVE_ASYNC_LDS 0
#endif

#if HAVE_ASYNC_LDS
// global -> LDS async copy of 16 bytes per lane (tracked by ASYNCcnt).
__device__ __forceinline__ void async_b128(const void* g, void* l) {
  __builtin_amdgcn_global_load_async_to_lds_b128(
      (AS1 v4i*)g, (AS3 v4i*)l, 0, 0);
}
#endif

#define SELU_L 1.0507009873554805f
#define SELU_A 1.6732632423543772f

// Branchless SELU: exp() runs on the TRANS pipe unconditionally (co-executes),
// result picked with v_cndmask instead of EXEC-mask branches.
__device__ __forceinline__ float selu_f(float x) {
  float e = __expf(fminf(x, 0.0f));
  float neg = SELU_A * (e - 1.0f);
  return SELU_L * ((x > 0.0f) ? x : neg);
}

__device__ __forceinline__ v8f wmma16(v16h a, v16h b, v8f c) {
  return __builtin_amdgcn_wmma_f32_16x16x32_f16(false, a, false, b, (short)0, c,
                                                false, false);
}

// A-fragment (16x32 f16, row-major source): lane<16 -> K in {0..15},
// lane>=16 -> K in {16..31}. p points at K=0 of this lane's row.
__device__ __forceinline__ v16h load_a_run(const _Float16* p, int khalf) {
  v16h a;
#pragma unroll
  for (int j = 0; j < 8; ++j) a[j] = p[2 * khalf + j];
#pragma unroll
  for (int j = 0; j < 8; ++j) a[8 + j] = p[2 * khalf + 8 + j];
  return a;
}
// 16 contiguous halves (used for packed B frags and [N][K]-row-major B).
__device__ __forceinline__ v16h load_16h(const _Float16* p) {
  v16h b;
#pragma unroll
  for (int j = 0; j < 16; ++j) b[j] = p[j];
  return b;
}

// ---------------------------------------------------------------------------
// Pack a fp32 weight matrix into WMMA B-fragment order (f16).
// B[k][n] = src[n*sn + k*sk + tap*st].  dst tile t=kt*NT+nt, per-lane 16 halves.
// ---------------------------------------------------------------------------
__global__ __launch_bounds__(256) void k_pack_b(const float* __restrict__ src,
                                                _Float16* __restrict__ dst,
                                                int KT, int NT, int sn, int sk,
                                                int st, int dstTapStride) {
  int tap = blockIdx.y;
  int total = KT * NT * 512;
  int tid = blockIdx.x * 256 + threadIdx.x;
  if (tid >= total) return;
  int j = tid & 15;
  int lane = (tid >> 4) & 31;
  int t = tid >> 9;
  int nt = t % NT, kt = t / NT;
  int k = kt * 32 + ((lane & 16) ? 16 : 0) + j;
  int n = nt * 16 + (lane & 15);
  dst[(size_t)tap * dstTapStride + tid] =
      (_Float16)src[(size_t)n * sn + (size_t)k * sk + (size_t)tap * st];
}

// ---------------------------------------------------------------------------
// K1: conv1(1->32, 6x4, SAME) + SELU + GN(per-channel) + maxpool 2x4
// block = (frame, oc); out A1[f][36][16][32] f16 (channel-last)
// ---------------------------------------------------------------------------
__global__ __launch_bounds__(256) void k1_conv1(const float* __restrict__ x,
    const float* __restrict__ w1, const float* __restrict__ b1,
    const float* __restrict__ g1, const float* __restrict__ gb1,
    _Float16* __restrict__ A1) {
  __shared__ float sIn[72 * 64];
  __shared__ float sOut[72 * 64];
  __shared__ float sRed[256];
  int f = blockIdx.x >> 5;
  int oc = blockIdx.x & 31;
  int tid = threadIdx.x;
  const float* xin = x + (size_t)f * 4608;
#if HAVE_ASYNC_LDS
  for (int c = tid; c < 1152; c += 256)
    async_b128(xin + c * 4, sIn + c * 4);
  asm volatile("s_wait_asynccnt 0" ::: "memory");
#else
  for (int i = tid; i < 4608; i += 256) sIn[i] = xin[i];
#endif
  float wk_[24];
#pragma unroll
  for (int i = 0; i < 24; ++i) wk_[i] = w1[oc * 24 + i];
  float bias = b1[oc];
  __syncthreads();
  float lsum = 0.0f, lss = 0.0f;
  for (int i = 0; i < 18; ++i) {
    int pix = tid + i * 256;
    int oh = pix >> 6, ow = pix & 63;
    float acc = bias;
#pragma unroll
    for (int kh = 0; kh < 6; ++kh) {
      int ih = oh + kh - 2;
      if (ih < 0 || ih >= 72) continue;
#pragma unroll
      for (int kw = 0; kw < 4; ++kw) {
        int iw = ow + kw - 1;
        if (iw < 0 || iw >= 64) continue;
        acc += sIn[ih * 64 + iw] * wk_[kh * 4 + kw];
      }
    }
    float y = selu_f(acc);
    sOut[pix] = y;
    lsum += y; lss += y * y;
  }
  sRed[tid] = lsum; __syncthreads();
  for (int s = 128; s > 0; s >>= 1) { if (tid < s) sRed[tid] += sRed[tid + s]; __syncthreads(); }
  float mean = sRed[0] * (1.0f / 4608.0f);
  __syncthreads();
  sRed[tid] = lss; __syncthreads();
  for (int s = 128; s > 0; s >>= 1) { if (tid < s) sRed[tid] += sRed[tid + s]; __syncthreads(); }
  float var = sRed[0] * (1.0f / 4608.0f) - mean * mean;
  float rs = rsqrtf(var + 1e-5f);
  float sc = g1[oc] * rs, sb = gb1[oc] - mean * sc;
  __syncthreads();
  _Float16* dst = A1 + (size_t)f * 36 * 16 * 32;
  for (int idx = tid; idx < 576; idx += 256) {
    int ph = idx >> 4, pw = idx & 15;
    float m = -1e30f;
#pragma unroll
    for (int dh = 0; dh < 2; ++dh)
#pragma unroll
      for (int dw = 0; dw < 4; ++dw)
        m = fmaxf(m, sOut[(ph * 2 + dh) * 64 + (pw * 4 + dw)] * sc + sb);
    dst[(ph * 16 + pw) * 32 + oc] = (_Float16)m;
  }
}

// ---------------------------------------------------------------------------
// K2: conv2(32->64, 6x4, SAME on 36x16) as 24 tap-GEMMs (M=576,N=64,K=32) via
// WMMA, fused SELU + GN(2ch/group) + maxpool 3x4. One workgroup per frame.
// A1 is channel-last, so the padded LDS tile fill is a contiguous row memcpy
// done with async global->LDS (16B per lane, ASYNCcnt tracked).
// ---------------------------------------------------------------------------
#define C2_PH 41
#define C2_PW 19
__global__ __launch_bounds__(256) void k2_conv2(const _Float16* __restrict__ A1,
    const _Float16* __restrict__ Wfrag, const float* __restrict__ b2,
    const float* __restrict__ g2, const float* __restrict__ gb2,
    _Float16* __restrict__ A2) {
  extern __shared__ char smem[];
  _Float16* inT = (_Float16*)smem;                       // 41*19*32 halves
  _Float16* outT = inT + C2_PH * C2_PW * 32;             // 64*576 halves
  float* sSum = (float*)(outT + 64 * 576);               // 64 + 64 floats
  float* sSS = sSum + 64;
  int f = blockIdx.x, tid = threadIdx.x;
  for (int i = tid; i < C2_PH * C2_PW * 32; i += 256) inT[i] = (_Float16)0.0f;
  if (tid < 128) sSum[tid] = 0.0f;  // covers sSum+sSS
  __syncthreads();
  const _Float16* src = A1 + (size_t)f * 36 * 16 * 32;
#if HAVE_ASYNC_LDS
  // 36 rows x 64 chunks of 16B (row = 16*32 halves = 1KB contiguous)
  for (int c = tid; c < 36 * 64; c += 256) {
    int h = c >> 6, seg = c & 63;
    async_b128(src + h * 512 + seg * 8,
               inT + ((h + 2) * C2_PW + 1) * 32 + seg * 8);
  }
  asm volatile("s_wait_asynccnt 0" ::: "memory");
#else
  for (int i = tid; i < 36 * 512; i += 256) {
    int h = i >> 9, rest = i & 511;
    inT[((h + 2) * C2_PW + 1) * 32 + rest] = src[h * 512 + rest];
  }
#endif
  __syncthreads();
  int wave = tid >> 5, lane = tid & 31;
  int r = lane & 15, hi = lane >> 4;
  for (int t = wave; t < 144; t += 8) {   // 36 Mtiles x 4 Ntiles
    int mt = t >> 2, nt = t & 3;
    int m = mt * 16 + r;
    int oh = m >> 4, ow = m & 15;
    v8f acc = {};
    for (int tap = 0; tap < 24; ++tap) {
      int kh = tap >> 2, kw = tap & 3;
      const _Float16* ap = &inT[((oh + kh) * C2_PW + (ow + kw)) * 32];
      v16h a = load_a_run(ap, hi ? 8 : 0);
      v16h b = load_16h(Wfrag + ((size_t)tap * 4 + nt) * 512 + lane * 16);
      acc = wmma16(a, b, acc);
    }
    int ocb = nt * 16 + (lane & 15);
    float bias = b2[ocb];
    float ls = 0.0f, lss = 0.0f;
#pragma unroll
    for (int rr = 0; rr < 8; ++rr) {
      int mm = mt * 16 + rr + hi * 8;
      float y = selu_f(acc[rr] + bias);
      outT[ocb * 576 + mm] = (_Float16)y;
      ls += y; lss += y * y;
    }
    atomicAdd(&sSum[ocb], ls);
    atomicAdd(&sSS[ocb], lss);
  }
  __syncthreads();
  _Float16* dst = A2 + (size_t)f * 12 * 4 * 64;   // channel-last [h][w][oc]
  for (int idx = tid; idx < 64 * 48; idx += 256) {
    int oc = idx / 48; int p = idx % 48; int ph = p >> 2, pw = p & 3;
    int g = oc >> 1;
    float sum = sSum[g * 2] + sSum[g * 2 + 1];
    float ss = sSS[g * 2] + sSS[g * 2 + 1];
    float mean = sum * (1.0f / 1152.0f);
    float var = ss * (1.0f / 1152.0f) - mean * mean;
    float rs = rsqrtf(var + 1e-5f);
    float sc = g2[oc] * rs, sb = gb2[oc] - mean * sc;
    float m = -1e30f;
#pragma unroll
    for (int dh = 0; dh < 3; ++dh)
#pragma unroll
      for (int dw = 0; dw < 4; ++dw)
        m = fmaxf(m, (float)outT[oc * 576 + (ph * 3 + dh) * 16 + (pw * 4 + dw)] * sc + sb);
    dst[(ph * 4 + pw) * 64 + oc] = (_Float16)m;
  }
}

// ---------------------------------------------------------------------------
// K3: conv3(64->128, 6x4, SAME on 12x4), 24 taps x (K=64 -> 2 chunks),
// M=48,N=128 via WMMA; SELU + GN(4ch/group) + maxpool 3x2; flatten -> Eflat.
// ---------------------------------------------------------------------------
#define C3_PH 17
#define C3_PW 7
__global__ __launch_bounds__(256) void k3_conv3(const _Float16* __restrict__ A2,
    const _Float16* __restrict__ Wfrag, const float* __restrict__ b3,
    const float* __restrict__ g3, const float* __restrict__ gb3,
    _Float16* __restrict__ Eflat) {
  extern __shared__ char smem[];
  _Float16* inT = (_Float16*)smem;                  // 17*7*64
  _Float16* outT = inT + C3_PH * C3_PW * 64;        // 128*48
  float* sSum = (float*)(outT + 128 * 48);
  float* sSS = sSum + 128;
  int f = blockIdx.x, tid = threadIdx.x;
  for (int i = tid; i < C3_PH * C3_PW * 64; i += 256) inT[i] = (_Float16)0.0f;
  if (tid < 128) sSum[tid] = 0.0f; else if (tid < 256) sSS[tid - 128] = 0.0f;
  __syncthreads();
  const _Float16* src = A2 + (size_t)f * 12 * 4 * 64;
#if HAVE_ASYNC_LDS
  // 12 rows x 32 chunks of 16B (row = 4*64 halves = 512B contiguous)
  for (int c = tid; c < 12 * 32; c += 256) {
    int h = c >> 5, seg = c & 31;
    async_b128(src + h * 256 + seg * 8,
               inT + ((h + 2) * C3_PW + 1) * 64 + seg * 8);
  }
  asm volatile("s_wait_asynccnt 0" ::: "memory");
#else
  for (int i = tid; i < 12 * 256; i += 256) {
    int h = i >> 8, rest = i & 255;
    inT[((h + 2) * C3_PW + 1) * 64 + rest] = src[h * 256 + rest];
  }
#endif
  __syncthreads();
  int wave = tid >> 5, lane = tid & 31;
  int r = lane & 15, hi = lane >> 4;
  int nt = wave;                    // 8 Ntiles
  for (int mt = 0; mt < 3; ++mt) {  // 3 Mtiles (M=48)
    int m = mt * 16 + r;
    int oh = m >> 2, ow = m & 3;
    v8f acc = {};
    for (int tap = 0; tap < 24; ++tap) {
      int kh = tap >> 2, kw = tap & 3;
      const _Float16* ap0 = &inT[((oh + kh) * C3_PW + (ow + kw)) * 64];
#pragma unroll
      for (int kt = 0; kt < 2; ++kt) {
        v16h a = load_a_run(ap0 + kt * 32, hi ? 8 : 0);
        v16h b = load_16h(Wfrag + ((size_t)tap * 16 + kt * 8 + nt) * 512 + lane * 16);
        acc = wmma16(a, b, acc);
      }
    }
    int ocb = nt * 16 + (lane & 15);
    float bias = b3[ocb];
    float ls = 0.0f, lss = 0.0f;
#pragma unroll
    for (int rr = 0; rr < 8; ++rr) {
      int mm = mt * 16 + rr + hi * 8;
      float y = selu_f(acc[rr] + bias);
      outT[ocb * 48 + mm] = (_Float16)y;
      ls += y; lss += y * y;
    }
    atomicAdd(&sSum[ocb], ls);
    atomicAdd(&sSS[ocb], lss);
  }
  __syncthreads();
  _Float16* dst = Eflat + (size_t)f * 1024;
  for (int idx = tid; idx < 128 * 8; idx += 256) {
    int oc = idx >> 3; int p = idx & 7; int ph = p >> 1, pw = p & 1;
    int g = oc >> 2;
    float sum = sSum[g * 4] + sSum[g * 4 + 1] + sSum[g * 4 + 2] + sSum[g * 4 + 3];
    float ss = sSS[g * 4] + sSS[g * 4 + 1] + sSS[g * 4 + 2] + sSS[g * 4 + 3];
    float mean = sum * (1.0f / 192.0f);
    float var = ss * (1.0f / 192.0f) - mean * mean;
    float rs = rsqrtf(var + 1e-5f);
    float sc = g3[oc] * rs, sb = gb3[oc] - mean * sc;
    float m = -1e30f;
#pragma unroll
    for (int dh = 0; dh < 3; ++dh)
#pragma unroll
      for (int dw = 0; dw < 2; ++dw)
        m = fmaxf(m, (float)outT[oc * 48 + (ph * 3 + dh) * 4 + (pw * 2 + dw)] * sc + sb);
    dst[oc * 8 + ph * 2 + pw] = (_Float16)m;
  }
}

// ---------------------------------------------------------------------------
// K4a: h1 = selu(Eflat @ fc1_w^T + b)   (M=1024, N=128, K=1024) WMMA
// ---------------------------------------------------------------------------
__global__ __launch_bounds__(256) void k4a_fc1(const _Float16* __restrict__ Eflat,
    const _Float16* __restrict__ Wfrag, const float* __restrict__ fb,
    float* __restrict__ h1) {
  int wave = threadIdx.x >> 5, lane = threadIdx.x & 31;
  int r = lane & 15, hi = lane >> 4;
  int mt = blockIdx.x, nt = wave;
  const _Float16* arow = Eflat + (size_t)(mt * 16 + r) * 1024;
  v8f acc = {};
  for (int kt = 0; kt < 32; ++kt) {
    v16h a = load_a_run(arow + kt * 32, hi ? 8 : 0);
    v16h b = load_16h(Wfrag + ((size_t)kt * 8 + nt) * 512 + lane * 16);
    acc = wmma16(a, b, acc);
  }
  int n = nt * 16 + (lane & 15);
  float bias = fb[n];
#pragma unroll
  for (int rr = 0; rr < 8; ++rr)
    h1[(size_t)(mt * 16 + rr + hi * 8) * 128 + n] = selu_f(acc[rr] + bias);
}

// K4b: per-row L2 normalize -> e (f32 + f16)
__global__ __launch_bounds__(128) void k4b_rownorm(const float* __restrict__ h1,
    float* __restrict__ e, _Float16* __restrict__ eh) {
  __shared__ float sr[128];
  int s = blockIdx.x, t = threadIdx.x;
  float v = h1[(size_t)s * 128 + t];
  sr[t] = v * v; __syncthreads();
  for (int k = 64; k > 0; k >>= 1) { if (t < k) sr[t] += sr[t + k]; __syncthreads(); }
  float inv = 1.0f / fmaxf(sqrtf(sr[0]), 1e-12f);
  float o = v * inv;
  e[(size_t)s * 128 + t] = o;
  eh[(size_t)s * 128 + t] = (_Float16)o;
}

// K4c: q/k/v = selu(e @ W^T + b); k gets positional encoding. WMMA.
__global__ __launch_bounds__(256) void k4c_qkv(const _Float16* __restrict__ eh,
    const _Float16* __restrict__ Wq, const _Float16* __restrict__ Wk,
    const _Float16* __restrict__ Wv, const float* __restrict__ bq,
    const float* __restrict__ bk, const float* __restrict__ bv,
    _Float16* __restrict__ qh, _Float16* __restrict__ kposh,
    float* __restrict__ vf) {
  int which = blockIdx.y;
  const _Float16* Wf = (which == 0) ? Wq : ((which == 1) ? Wk : Wv);
  const float* bb = (which == 0) ? bq : ((which == 1) ? bk : bv);
  int wave = threadIdx.x >> 5, lane = threadIdx.x & 31;
  int r = lane & 15, hi = lane >> 4;
  int mt = blockIdx.x, nt = wave;
  const _Float16* arow = eh + (size_t)(mt * 16 + r) * 128;
  v8f acc = {};
#pragma unroll
  for (int kt = 0; kt < 4; ++kt) {
    v16h a = load_a_run(arow + kt * 32, hi ? 8 : 0);
    v16h b = load_16h(Wf + ((size_t)kt * 8 + nt) * 512 + lane * 16);
    acc = wmma16(a, b, acc);
  }
  int n = nt * 16 + (lane & 15);
  float bias = bb[n];
#pragma unroll
  for (int rr = 0; rr < 8; ++rr) {
    int s = mt * 16 + rr + hi * 8;
    float y = selu_f(acc[rr] + bias);
    if (which == 0) {
      qh[(size_t)s * 128 + n] = (_Float16)y;
    } else if (which == 1) {
      int i2 = (n >> 1) * 2;
      float div = __expf(-9.210340371976184f * (float)i2 / 128.0f);
      float arg = (float)s * div;
      y += (n & 1) ? __cosf(arg) : __sinf(arg);
      kposh[(size_t)s * 128 + n] = (_Float16)y;
    } else {
      vf[(size_t)s * 128 + n] = y;
    }
  }
}

// K5a: scores = (q @ kpos^T) / sqrt(128)   (M=N=1024, K=128) WMMA
__global__ __launch_bounds__(256) void k5a_scores(const _Float16* __restrict__ qh,
    const _Float16* __restrict__ kposh, float* __restrict__ scores) {
  int wave = threadIdx.x >> 5, lane = threadIdx.x & 31;
  int r = lane & 15, hi = lane >> 4;
  int mt = blockIdx.x, nt = blockIdx.y * 8 + wave;
  const _Float16* arow = qh + (size_t)(mt * 16 + r) * 128;
  const _Float16* brow = kposh + (size_t)(nt * 16 + (lane & 15)) * 128 + (hi ? 16 : 0);
  v8f acc = {};
#pragma unroll
  for (int kt = 0; kt < 4; ++kt) {
    v16h a = load_a_run(arow + kt * 32, hi ? 8 : 0);
    v16h b = load_16h(brow + kt * 32);
    acc = wmma16(a, b, acc);
  }
  int n = nt * 16 + (lane & 15);
#pragma unroll
  for (int rr = 0; rr < 8; ++rr)
    scores[(size_t)(mt * 16 + rr + hi * 8) * 1024 + n] = acc[rr] * 0.08838834764831843f;
}

// K5b: row softmax -> f16 probs
__global__ __launch_bounds__(256) void k5b_softmax(const float* __restrict__ scores,
    _Float16* __restrict__ attnh) {
  __shared__ float sr[256];
  int row = blockIdx.x, t = threadIdx.x;
  float lv[4];
#pragma unroll
  for (int i = 0; i < 4; ++i) lv[i] = scores[(size_t)row * 1024 + t + i * 256];
  float mx = fmaxf(fmaxf(lv[0], lv[1]), fmaxf(lv[2], lv[3]));
  sr[t] = mx; __syncthreads();
  for (int k = 128; k > 0; k >>= 1) { if (t < k) sr[t] = fmaxf(sr[t], sr[t + k]); __syncthreads(); }
  mx = sr[0]; __syncthreads();
  float ls = 0.0f;
#pragma unroll
  for (int i = 0; i < 4; ++i) { lv[i] = __expf(lv[i] - mx); ls += lv[i]; }
  sr[t] = ls; __syncthreads();
  for (int k = 128; k > 0; k >>= 1) { if (t < k) sr[t] += sr[t + k]; __syncthreads(); }
  float inv = 1.0f / sr[0];
#pragma unroll
  for (int i = 0; i < 4; ++i)
    attnh[(size_t)row * 1024 + t + i * 256] = (_Float16)(lv[i] * inv);
}

// K5c: out = attn @ v   (M=1024, N=128, K=1024) WMMA, v prepacked as B frags
__global__ __launch_bounds__(256) void k5c_av(const _Float16* __restrict__ attnh,
    const _Float16* __restrict__ Vfrag, float* __restrict__ outf) {
  int wave = threadIdx.x >> 5, lane = threadIdx.x & 31;
  int r = lane & 15, hi = lane >> 4;
  int mt = blockIdx.x, nt = wave;
  const _Float16* arow = attnh + (size_t)(mt * 16 + r) * 1024;
  v8f acc = {};
  for (int kt = 0; kt < 32; ++kt) {
    v16h a = load_a_run(arow + kt * 32, hi ? 8 : 0);
    v16h b = load_16h(Vfrag + ((size_t)kt * 8 + nt) * 512 + lane * 16);
    acc = wmma16(a, b, acc);
  }
  int n = nt * 16 + (lane & 15);
#pragma unroll
  for (int rr = 0; rr < 8; ++rr)
    outf[(size_t)(mt * 16 + rr + hi * 8) * 128 + n] = acc[rr];
}

// column (axis=0) L2 norms of a [1024,128] matrix
__global__ __launch_bounds__(256) void k_colnorm(const float* __restrict__ m,
                                                 float* __restrict__ cn) {
  __shared__ float sr[256];
  int j = blockIdx.x, t = threadIdx.x;
  float a = 0.0f;
  for (int s = t; s < 1024; s += 256) { float v = m[(size_t)s * 128 + j]; a += v * v; }
  sr[t] = a; __syncthreads();
  for (int k = 128; k > 0; k >>= 1) { if (t < k) sr[t] += sr[t + k]; __syncthreads(); }
  if (t == 0) cn[j] = fmaxf(sqrtf(sr[0]), 1e-12f);
}

// tmp = out/cn1 + e
__global__ __launch_bounds__(256) void k5e_add(const float* __restrict__ outf,
    const float* __restrict__ cn1, const float* __restrict__ e,
    float* __restrict__ tmp) {
  int i = blockIdx.x * 256 + threadIdx.x;
  int j = i & 127;
  tmp[i] = outf[i] / cn1[j] + e[i];
}

// ea = tmp/cn2 (f32 + f16)
__global__ __launch_bounds__(256) void k5g_scale(const float* __restrict__ tmp,
    const float* __restrict__ cn2, float* __restrict__ eaf,
    _Float16* __restrict__ eah) {
  int i = blockIdx.x * 256 + threadIdx.x;
  int j = i & 127;
  float v = tmp[i] / cn2[j];
  eaf[i] = v;
  eah[i] = (_Float16)v;
}

// sq[s] = sum_j ea[s][j]^2
__global__ __launch_bounds__(128) void k5h_rowsq(const float* __restrict__ eaf,
                                                 float* __restrict__ sq) {
  __shared__ float sr[128];
  int s = blockIdx.x, t = threadIdx.x;
  float v = eaf[(size_t)s * 128 + t];
  sr[t] = v * v; __syncthreads();
  for (int k = 64; k > 0; k >>= 1) { if (t < k) sr[t] += sr[t + k]; __syncthreads(); }
  if (t == 0) sq[s] = sr[0];
}

__global__ void k_init(unsigned int* dmax) { *dmax = 0u; }

// K6: G = ea@ea^T via WMMA; d = sqrt(max(sqi+sqj-2G,0)+1e-12) -> d_out; track max
__global__ __launch_bounds__(256) void k6_dist(const _Float16* __restrict__ eah,
    const float* __restrict__ sq, float* __restrict__ dmat,
    unsigned int* __restrict__ dmax) {
  __shared__ float sr[256];
  int wave = threadIdx.x >> 5, lane = threadIdx.x & 31;
  int r = lane & 15, hi = lane >> 4;
  int mt = blockIdx.x, nt = blockIdx.y * 8 + wave;
  const _Float16* arow = eah + (size_t)(mt * 16 + r) * 128;
  const _Float16* brow = eah + (size_t)(nt * 16 + (lane & 15)) * 128 + (hi ? 16 : 0);
  v8f acc = {};
#pragma unroll
  for (int kt = 0; kt < 4; ++kt) {
    v16h a = load_a_run(arow + kt * 32, hi ? 8 : 0);
    v16h b = load_16h(brow + kt * 32);
    acc = wmma16(a, b, acc);
  }
  int jcol = nt * 16 + (lane & 15);
  float sqj = sq[jcol];
  float lmax = 0.0f;
#pragma unroll
  for (int rr = 0; rr < 8; ++rr) {
    int irow = mt * 16 + rr + hi * 8;
    float d2 = sq[irow] + sqj - 2.0f * acc[rr];
    float d = sqrtf(fmaxf(d2, 0.0f) + 1e-12f);
    dmat[(size_t)irow * 1024 + jcol] = d;
    lmax = fmaxf(lmax, d);
  }
  sr[threadIdx.x] = lmax; __syncthreads();
  for (int k = 128; k > 0; k >>= 1) { if (threadIdx.x < k) sr[threadIdx.x] = fmaxf(sr[threadIdx.x], sr[threadIdx.x + k]); __syncthreads(); }
  if (threadIdx.x == 0) atomicMax(dmax, __float_as_uint(sr[0]));
}

__global__ __launch_bounds__(256) void k7_final(float* __restrict__ dmat,
                                                const unsigned int* __restrict__ dmax) {
  float inv = 1.0f / __uint_as_float(*dmax);
  int i = blockIdx.x * 256 + threadIdx.x;
  dmat[i] = 1.0f - dmat[i] * inv;
}

// ---------------------------------------------------------------------------
extern "C" void kernel_launch(void* const* d_in, const int* in_sizes, int n_in,
                              void* d_out, int out_size, void* d_ws, size_t ws_size,
                              hipStream_t stream) {
  (void)in_sizes; (void)n_in; (void)out_size; (void)ws_size;
  const float* x    = (const float*)d_in[0];
  const float* c1w  = (const float*)d_in[1];
  const float* c1b  = (const float*)d_in[2];
  const float* g1g  = (const float*)d_in[3];
  const float* g1b  = (const float*)d_in[4];
  const float* c2w  = (const float*)d_in[5];
  const float* c2b  = (const float*)d_in[6];
  const float* g2g  = (const float*)d_in[7];
  const float* g2b  = (const float*)d_in[8];
  const float* c3w  = (const float*)d_in[9];
  const float* c3b  = (const float*)d_in[10];
  const float* g3g  = (const float*)d_in[11];
  const float* g3b  = (const float*)d_in[12];
  const float* fc1w = (const float*)d_in[13];
  const float* fc1b = (const float*)d_in[14];
  const float* wq   = (const float*)d_in[15];
  const float* bq   = (const float*)d_in[16];
  const float* wk   = (const float*)d_in[17];
  const float* bk   = (const float*)d_in[18];
  const float* wv   = (const float*)d_in[19];
  const float* bv   = (const float*)d_in[20];
  float* dout = (float*)d_out;

  char* base = (char*)d_ws;
  size_t off = 0;
  auto alloc = [&](size_t bytes) -> void* {
    size_t a = (off + 255) & ~(size_t)255;
    void* r = base + a;
    off = a + bytes;
    return r;
  };
  _Float16* A1    = (_Float16*)alloc((size_t)1024 * 32 * 576 * 2);
  _Float16* A2    = (_Float16*)alloc((size_t)1024 * 64 * 48 * 2);
  _Float16* Eflat = (_Float16*)alloc((size_t)1024 * 1024 * 2);
  _Float16* Wf2   = (_Float16*)alloc((size_t)24 * 2048 * 2);
  _Float16* Wf3   = (_Float16*)alloc((size_t)24 * 8192 * 2);
  _Float16* WfF   = (_Float16*)alloc((size_t)32 * 8 * 512 * 2);
  _Float16* WfQ   = (_Float16*)alloc((size_t)4 * 8 * 512 * 2);
  _Float16* WfK   = (_Float16*)alloc((size_t)4 * 8 * 512 * 2);
  _Float16* WfV   = (_Float16*)alloc((size_t)4 * 8 * 512 * 2);
  _Float16* VfB   = (_Float16*)alloc((size_t)32 * 8 * 512 * 2);
  float*    h1    = (float*)alloc((size_t)1024 * 128 * 4);
  float*    eF    = (float*)alloc((size_t)1024 * 128 * 4);
  _Float16* eH    = (_Float16*)alloc((size_t)1024 * 128 * 2);
  _Float16* qH    = (_Float16*)alloc((size_t)1024 * 128 * 2);
  _Float16* kH    = (_Float16*)alloc((size_t)1024 * 128 * 2);
  float*    vF    = (float*)alloc((size_t)1024 * 128 * 4);
  float*    scores= (float*)alloc((size_t)1024 * 1024 * 4);
  _Float16* attnH = (_Float16*)alloc((size_t)1024 * 1024 * 2);
  float*    outF  = (float*)alloc((size_t)1024 * 128 * 4);
  float*    tmpF  = (float*)alloc((size_t)1024 * 128 * 4);
  float*    eaF   = (float*)alloc((size_t)1024 * 128 * 4);
  _Float16* eaH   = (_Float16*)alloc((size_t)1024 * 128 * 2);
  float*    cn1   = (float*)alloc(128 * 4);
  float*    cn2   = (float*)alloc(128 * 4);
  float*    sq    = (float*)alloc(1024 * 4);
  unsigned int* dmax = (unsigned int*)alloc(4);

  // weight fragment packing (L2-resident, reused by all 1024 frames)
  k_pack_b<<<dim3(8, 24), 256, 0, stream>>>(c2w, Wf2, 1, 4, 768, 24, 1, 2048);
  k_pack_b<<<dim3(32, 24), 256, 0, stream>>>(c3w, Wf3, 2, 8, 1536, 24, 1, 8192);
  k_pack_b<<<dim3(512, 1), 256, 0, stream>>>(fc1w, WfF, 32, 8, 1024, 1, 0, 0);
  k_pack_b<<<dim3(64, 1), 256, 0, stream>>>(wq, WfQ, 4, 8, 128, 1, 0, 0);
  k_pack_b<<<dim3(64, 1), 256, 0, stream>>>(wk, WfK, 4, 8, 128, 1, 0, 0);
  k_pack_b<<<dim3(64, 1), 256, 0, stream>>>(wv, WfV, 4, 8, 128, 1, 0, 0);
  k_init<<<1, 1, 0, stream>>>(dmax);

  k1_conv1<<<32768, 256, 0, stream>>>(x, c1w, c1b, g1g, g1b, A1);
  size_t smem2 = (size_t)(C2_PH * C2_PW * 32 + 64 * 576) * 2 + 128 * 4;
  k2_conv2<<<1024, 256, smem2, stream>>>(A1, Wf2, c2b, g2g, g2b, A2);
  size_t smem3 = (size_t)(C3_PH * C3_PW * 64 + 128 * 48) * 2 + 256 * 4;
  k3_conv3<<<1024, 256, smem3, stream>>>(A2, Wf3, c3b, g3g, g3b, Eflat);

  k4a_fc1<<<64, 256, 0, stream>>>(Eflat, WfF, fc1b, h1);
  k4b_rownorm<<<1024, 128, 0, stream>>>(h1, eF, eH);
  k4c_qkv<<<dim3(64, 3), 256, 0, stream>>>(eH, WfQ, WfK, WfV, bq, bk, bv, qH, kH, vF);
  // pack v (B[k=s][n=feat] = vF[k*128+n]) for attn@v
  k_pack_b<<<dim3(512, 1), 256, 0, stream>>>(vF, VfB, 32, 8, 1, 128, 0, 0);

  k5a_scores<<<dim3(64, 8), 256, 0, stream>>>(qH, kH, scores);
  k5b_softmax<<<1024, 256, 0, stream>>>(scores, attnH);
  k5c_av<<<64, 256, 0, stream>>>(attnH, VfB, outF);

  k_colnorm<<<128, 256, 0, stream>>>(outF, cn1);
  k5e_add<<<512, 256, 0, stream>>>(outF, cn1, eF, tmpF);
  k_colnorm<<<128, 256, 0, stream>>>(tmpF, cn2);
  k5g_scale<<<512, 256, 0, stream>>>(tmpF, cn2, eaF, eaH);
  k5h_rowsq<<<1024, 128, 0, stream>>>(eaF, sq);

  k6_dist<<<dim3(64, 8), 256, 0, stream>>>(eaH, sq, dout, dmax);
  k7_final<<<4096, 256, 0, stream>>>(dout, dmax);
}
